// FP8DynamicLinear_14912126451853
// MI455X (gfx1250) — 8-bit matrix operand order repaired and checked by reading, compiled, not run on hardware
//
#include <hip/hip_runtime.h>
#include <hip/hip_bf16.h>

typedef __attribute__((ext_vector_type(16))) int   v16i;
typedef __attribute__((ext_vector_type(8)))  float v8f;
typedef __attribute__((ext_vector_type(4)))  int   v4i_t;

#if __has_builtin(__builtin_amdgcn_global_load_async_to_lds_b128)
#define HAVE_ASYNC_LDS 1
#endif

#ifdef HAVE_ASYNC_LDS
#if __has_builtin(__builtin_amdgcn_s_wait_asynccnt)
#define WAIT_ASYNC(n) __builtin_amdgcn_s_wait_asynccnt(n)
#else
#define WAIT_ASYNC(n) asm volatile("s_wait_asynccnt %0" ::"i"(n) : "memory")
#endif
#else
#define WAIT_ASYNC(n) ((void)0)
#endif

#define AS1 __attribute__((address_space(1)))
#define AS3 __attribute__((address_space(3)))

__device__ __forceinline__ unsigned char f32_to_e4m3_sw(float x) {
    float ax = fabsf(x);
    unsigned char s = (x < 0.0f) ? 0x80 : 0x00;
    ax = fminf(ax, 448.0f);
    if (ax < 0.0009765625f) return s;
    int e;
    float m = frexpf(ax, &e);
    int E = e + 6;
    if (E <= 0) {
        unsigned int mant = (unsigned int)rintf(ax * 512.0f);
        if (mant >= 8) return s | 0x08;
        return s | (unsigned char)mant;
    }
    unsigned int mant = (unsigned int)rintf(m * 16.0f) - 8u;
    if (mant == 8u) { mant = 0u; E += 1; }
    if (E > 15 || (E == 15 && mant > 6u)) { E = 15; mant = 6u; }
    return s | (unsigned char)((E << 3) | mant);
}

__device__ __forceinline__ unsigned int pack4_e4m3(float a, float b, float c, float d) {
#if __has_builtin(__builtin_amdgcn_cvt_pk_fp8_f32)
    int p = 0;
    p = __builtin_amdgcn_cvt_pk_fp8_f32(a, b, p, false);
    p = __builtin_amdgcn_cvt_pk_fp8_f32(c, d, p, true);
    return (unsigned int)p;
#else
    return (unsigned int)f32_to_e4m3_sw(a)
         | ((unsigned int)f32_to_e4m3_sw(b) << 8)
         | ((unsigned int)f32_to_e4m3_sw(c) << 16)
         | ((unsigned int)f32_to_e4m3_sw(d) << 24);
#endif
}

__device__ __forceinline__ float bf16_round(float f) {
    return __bfloat162float(__float2bfloat16(f));
}

__device__ __forceinline__ unsigned int pack_bf16x2(float a, float b) {
    __hip_bfloat16 ha = __float2bfloat16(a);
    __hip_bfloat16 hb = __float2bfloat16(b);
    unsigned short ua, ub;
    __builtin_memcpy(&ua, &ha, 2);
    __builtin_memcpy(&ub, &hb, 2);
    return ((unsigned int)ub << 16) | (unsigned int)ua;
}

__global__ void init_amax_kernel(unsigned int* amax_bits) {
    if (threadIdx.x == 0) *amax_bits = 0u;
}

__global__ void amax_kernel(const float4* __restrict__ x4, long long n4,
                            unsigned int* __restrict__ amax_bits) {
    float m = 0.0f;
    long long stride = (long long)gridDim.x * blockDim.x;
    for (long long i = (long long)blockIdx.x * blockDim.x + threadIdx.x; i < n4; i += stride) {
        float4 v = x4[i];
        m = fmaxf(m, fabsf(bf16_round(v.x)));
        m = fmaxf(m, fabsf(bf16_round(v.y)));
        m = fmaxf(m, fabsf(bf16_round(v.z)));
        m = fmaxf(m, fabsf(bf16_round(v.w)));
    }
    #pragma unroll
    for (int off = 16; off > 0; off >>= 1)
        m = fmaxf(m, __shfl_xor(m, off, 32));
    __shared__ float red[8];
    int wave = threadIdx.x >> 5;
    int lane = threadIdx.x & 31;
    if (lane == 0) red[wave] = m;
    __syncthreads();
    if (threadIdx.x == 0) {
        float b = red[0];
        #pragma unroll
        for (int i = 1; i < 8; ++i) b = fmaxf(b, red[i]);
        atomicMax(amax_bits, __float_as_uint(b));
    }
}

__global__ void quant_x_kernel(const float4* __restrict__ x4, long long n4,
                               const unsigned int* __restrict__ amax_bits,
                               unsigned int* __restrict__ qx) {
    float amax  = __uint_as_float(*amax_bits);
    float scale = 448.0f / fmaxf(amax, 1e-12f);
    long long stride = (long long)gridDim.x * blockDim.x;
    for (long long i = (long long)blockIdx.x * blockDim.x + threadIdx.x; i < n4; i += stride) {
        float4 v = x4[i];
        float a = fminf(fmaxf(bf16_round(v.x) * scale, -448.0f), 448.0f);
        float b = fminf(fmaxf(bf16_round(v.y) * scale, -448.0f), 448.0f);
        float c = fminf(fmaxf(bf16_round(v.z) * scale, -448.0f), 448.0f);
        float d = fminf(fmaxf(bf16_round(v.w) * scale, -448.0f), 448.0f);
        qx[i] = pack4_e4m3(a, b, c, d);
    }
}

__global__ void quant_w_kernel(const float4* __restrict__ w4, long long n4,
                               unsigned int* __restrict__ qw) {
    long long stride = (long long)gridDim.x * blockDim.x;
    for (long long i = (long long)blockIdx.x * blockDim.x + threadIdx.x; i < n4; i += stride) {
        float4 v = w4[i];
        float a = fminf(fmaxf(v.x, -448.0f), 448.0f);
        float b = fminf(fmaxf(v.y, -448.0f), 448.0f);
        float c = fminf(fmaxf(v.z, -448.0f), 448.0f);
        float d = fminf(fmaxf(v.w, -448.0f), 448.0f);
        qw[i] = pack4_e4m3(a, b, c, d);
    }
}

#define BK        128
#define LDS_STRIDE 144
#define A_TILE_B  (64  * LDS_STRIDE)
#define B_TILE_B  (128 * LDS_STRIDE)
#define BUF_B     (A_TILE_B + B_TILE_B)
#define EPI_STRIDE 36

__global__ void __launch_bounds__(256, 2)
gemm_fp8_kernel(const unsigned char* __restrict__ qx,
                const unsigned char* __restrict__ qw,
                const unsigned int* __restrict__ amax_bits,
                const float* __restrict__ wscale,
                const __hip_bfloat16* __restrict__ bias,
                __hip_bfloat16* __restrict__ out,
                int M, int N, int K) {
    __shared__ __align__(16) char smem[2 * BUF_B];

    const int t    = threadIdx.x;
    const int lane = t & 31;
    const int wave = t >> 5;
    const int half = lane >> 4;
    const int l16  = lane & 15;

    const int wm  = wave >> 2;
    const int wn  = wave & 3;
    const int gmB = blockIdx.y * 64;
    const int gnB = blockIdx.x * 128;
    const int gm0 = gmB + wm * 32;
    const int gn0 = gnB + wn * 32;

    v8f acc[2][2];
    #pragma unroll
    for (int i = 0; i < 2; ++i)
        #pragma unroll
        for (int j = 0; j < 2; ++j)
            acc[i][j] = (v8f){0.f, 0.f, 0.f, 0.f, 0.f, 0.f, 0.f, 0.f};

    auto issue_tile = [&](int buf, int kb) {
        char* bufA = smem + buf * BUF_B;
        char* bufB = bufA + A_TILE_B;
        #pragma unroll
        for (int i = 0; i < 2; ++i) {
            int c   = t + 256 * i;
            int row = c >> 3;
            int off = (c & 7) << 4;
            const unsigned char* g = qx + (size_t)(gmB + row) * K + kb + off;
            char* l = bufA + row * LDS_STRIDE + off;
#ifdef HAVE_ASYNC_LDS
            __builtin_amdgcn_global_load_async_to_lds_b128(
                (AS1 v4i_t*)g, (AS3 v4i_t*)l, 0, 0);
#else
            *(int4*)l = *(const int4*)g;
#endif
        }
        #pragma unroll
        for (int i = 0; i < 4; ++i) {
            int c   = t + 256 * i;
            int row = c >> 3;
            int off = (c & 7) << 4;
            const unsigned char* g = qw + (size_t)(gnB + row) * K + kb + off;
            char* l = bufB + row * LDS_STRIDE + off;
#ifdef HAVE_ASYNC_LDS
            __builtin_amdgcn_global_load_async_to_lds_b128(
                (AS1 v4i_t*)g, (AS3 v4i_t*)l, 0, 0);
#else
            *(int4*)l = *(const int4*)g;
#endif
        }
    };

    const int nsteps = K >> 7;
    issue_tile(0, 0);

    for (int it = 0; it < nsteps; ++it) {
        if (it + 1 < nsteps) {
            issue_tile((it + 1) & 1, (it + 1) * BK);
            WAIT_ASYNC(6);
        } else {
            WAIT_ASYNC(0);
        }
        __syncthreads();

        const char* bufA = smem + (it & 1) * BUF_B;
        const char* bufB = bufA + A_TILE_B;

        v16i af[2], bf[2];
        #pragma unroll
        for (int ti = 0; ti < 2; ++ti) {
            const char* p = bufA + (wm * 32 + 16 * ti + l16) * LDS_STRIDE + 8 * half;
            #pragma unroll
            for (int i = 0; i < 8; ++i) {
                int2 v = *(const int2*)(p + 16 * i);
                af[ti][2 * i]     = v.x;
                af[ti][2 * i + 1] = v.y;
            }
        }
        #pragma unroll
        for (int tj = 0; tj < 2; ++tj) {
            const char* p = bufB + (wn * 32 + 16 * tj + l16) * LDS_STRIDE + 8 * half;
            #pragma unroll
            for (int j = 0; j < 8; ++j) {
                int2 v = *(const int2*)(p + 16 * j);
                bf[tj][2 * j]     = v.x;
                bf[tj][2 * j + 1] = v.y;
            }
        }
        #pragma unroll
        for (int ti = 0; ti < 2; ++ti)
            #pragma unroll
            for (int tj = 0; tj < 2; ++tj)
                acc[ti][tj] = __builtin_amdgcn_wmma_f32_16x16x128_fp8_fp8(
                    af[ti], bf[tj], (short)0, acc[ti][tj],
                     false,  false);

        __syncthreads();
    }

    const float xs   = fmaxf(__uint_as_float(*amax_bits), 1e-12f) * (1.0f / 448.0f);
    const float comb = xs * wscale[0];

    float* wt = (float*)smem + wave * (32 * EPI_STRIDE);
    #pragma unroll
    for (int ti = 0; ti < 2; ++ti) {
        #pragma unroll
        for (int tj = 0; tj < 2; ++tj) {
            const int n  = gn0 + 16 * tj + l16;
            const float bv = __bfloat162float(bias[n]);
            #pragma unroll
            for (int v = 0; v < 8; ++v) {
                const int r = 16 * ti + v + 8 * half;
                wt[r * EPI_STRIDE + 16 * tj + l16] = acc[ti][tj][v] * comb + bv;
            }
        }
    }
    __syncthreads();

    {
        const float* src = wt + lane * EPI_STRIDE;
        __hip_bfloat16* dstb = out + (size_t)(gm0 + lane) * N + gn0;
        #pragma unroll
        for (int j = 0; j < 4; ++j) {
            float4 f0 = *(const float4*)(src + 8 * j);
            float4 f1 = *(const float4*)(src + 8 * j + 4);
            uint4 o;
            o.x = pack_bf16x2(f0.x, f0.y);
            o.y = pack_bf16x2(f0.z, f0.w);
            o.z = pack_bf16x2(f1.x, f1.y);
            o.w = pack_bf16x2(f1.z, f1.w);
            *(uint4*)(dstb + 8 * j) = o;
        }
    }
}

extern "C" void kernel_launch(void* const* d_in, const int* in_sizes, int n_in,
                              void* d_out, int out_size, void* d_ws, size_t ws_size,
                              hipStream_t stream) {
    const float*          x      = (const float*)d_in[0];
    const float*          weight = (const float*)d_in[1];
    const float*          wscale = (const float*)d_in[2];
    const __hip_bfloat16* bias   = (const __hip_bfloat16*)d_in[3];
    __hip_bfloat16*       out    = (__hip_bfloat16*)d_out;

    const long long N = in_sizes[3];
    const long long K = (long long)in_sizes[1] / N;
    const long long M = (long long)in_sizes[0] / K;

    unsigned int*  amax_bits = (unsigned int*)d_ws;
    unsigned char* qx        = (unsigned char*)d_ws + 256;
    unsigned char* qw        = qx + (size_t)M * K;

    init_amax_kernel<<<1, 64, 0, stream>>>(amax_bits);

    const long long nx4 = M * K / 4;
    const long long nw4 = N * K / 4;
    amax_kernel<<<2048, 256, 0, stream>>>((const float4*)x, nx4, amax_bits);
    quant_x_kernel<<<2048, 256, 0, stream>>>((const float4*)x, nx4, amax_bits,
                                             (unsigned int*)qx);
    quant_w_kernel<<<4096, 256, 0, stream>>>((const float4*)weight, nw4,
                                             (unsigned int*)qw);

    dim3 grid((unsigned)(N / 128), (unsigned)(M / 64));
    gemm_fp8_kernel<<<grid, 256, 0, stream>>>(qx, qw, amax_bits, wscale, bias,
                                              out, (int)M, (int)N, (int)K);
}
